// MultiHeadSelfAttention_3246995276232
// MI455X (gfx1250) — matrix operand order repaired and checked by reading, not compiled, not run on hardware
//
#include <hip/hip_runtime.h>

typedef __bf16 bf16;
typedef __attribute__((ext_vector_type(16))) __bf16 v16bf;
typedef __attribute__((ext_vector_type(8)))  float  v8f;

union FragU { v16bf f; uint4 u[2]; };

// ---------------------------------------------------------------------------
// CDNA5 async global->LDS copy (ASYNCcnt path), with portable fallback.
// Builtin signature (from hipcc diagnostic): (v4i AS1*, v4i AS3*, Ii, Ii).
// ---------------------------------------------------------------------------
#if defined(__gfx1250__) && __has_builtin(__builtin_amdgcn_global_load_async_to_lds_b128)
#define HAVE_ASYNC_LDS 1
#endif

typedef int v4i __attribute__((vector_size(16)));
#ifdef HAVE_ASYNC_LDS
typedef __attribute__((address_space(1))) v4i v4i_g;
typedef __attribute__((address_space(3))) v4i v4i_l;
#endif

__device__ __forceinline__ void async_copy16(const void* gsrc, void* lds_dst) {
#ifdef HAVE_ASYNC_LDS
  __builtin_amdgcn_global_load_async_to_lds_b128(
      (v4i_g*)(void*)gsrc, (v4i_l*)lds_dst, 0, 0);
#else
  *reinterpret_cast<uint4*>(lds_dst) = *reinterpret_cast<const uint4*>(gsrc);
#endif
}

__device__ __forceinline__ void async_drain() {
#ifdef HAVE_ASYNC_LDS
  asm volatile("s_wait_asynccnt 0" ::: "memory");
#endif
}

// ---------------------------------------------------------------------------
// WMMA helpers (CDNA5: D = A(16x32) * B(32x16) + C, bf16 in / f32 acc)
// ---------------------------------------------------------------------------
__device__ __forceinline__ v8f wmma_bf16(v16bf a, v16bf b, v8f c) {
  return __builtin_amdgcn_wmma_f32_16x16x32_bf16(
      /*neg_a=*/false, a, /*neg_b=*/false, b,
      /*c_mod=*/(short)0, c, /*reuse_a=*/false, /*reuse_b=*/false);
}

// A-operand (16x32, M x K) from row-major [M x ld] bf16.
// Lane L(0-15): row L, K = [k0+half*8, +8) and [k0+16+half*8, +8).
__device__ __forceinline__ v16bf load_a(const bf16* base, int ld, int row0, int k0) {
  int lane = threadIdx.x & 31;
  int m = lane & 15, half = lane >> 4;
  const bf16* p0 = base + (size_t)(row0 + m) * ld + k0 + half * 8;
  FragU fu;
  fu.u[0] = *reinterpret_cast<const uint4*>(p0);
  fu.u[1] = *reinterpret_cast<const uint4*>(p0 + 16);
  return fu.f;
}

// B-operand (32x16, K x N) from row-major [N x ld] bf16 (i.e. B transposed).
// Lane n(0-15): col n, K = [k0+half*8, +8) and [k0+16+half*8, +8).
__device__ __forceinline__ v16bf load_b(const bf16* base, int ld, int n0, int k0) {
  int lane = threadIdx.x & 31;
  int n = lane & 15, half = lane >> 4;
  const bf16* p0 = base + (size_t)(n0 + n) * ld + k0 + half * 8;
  FragU fu;
  fu.u[0] = *reinterpret_cast<const uint4*>(p0);
  fu.u[1] = *reinterpret_cast<const uint4*>(p0 + 16);
  return fu.f;
}

// ---------------------------------------------------------------------------
// Constants
// ---------------------------------------------------------------------------
#define BATCH 4
#define SEQ   2048
#define EMB   1024
#define HEADS 16
#define HDIM  64
#define MROWS (BATCH * SEQ)   // 8192

// ---------------------------------------------------------------------------
// Prep: X fp32 -> bf16; weights fp32 [K,N] -> bf16 [N,K] (transposed)
// ---------------------------------------------------------------------------
__global__ void prep_kernel(const float* __restrict__ X,
                            const float* __restrict__ wq, const float* __restrict__ wk,
                            const float* __restrict__ wv, const float* __restrict__ wo,
                            bf16* __restrict__ Xb,
                            bf16* __restrict__ wqT, bf16* __restrict__ wkT,
                            bf16* __restrict__ wvT, bf16* __restrict__ woT) {
  size_t idx = (size_t)blockIdx.x * blockDim.x + threadIdx.x;
  if (idx < (size_t)MROWS * EMB) Xb[idx] = (bf16)X[idx];
  if (idx < (size_t)EMB * EMB) {
    int n = (int)(idx >> 10), k = (int)(idx & 1023);
    size_t src = (size_t)k * EMB + n;
    wqT[idx] = (bf16)wq[src];
    wkT[idx] = (bf16)wk[src];
    wvT[idx] = (bf16)wv[src];
    woT[idx] = (bf16)wo[src];
  }
}

// ---------------------------------------------------------------------------
// GEMM: C[8192 x 1024] = A[8192 x 1024] * W + bias, W given transposed [N,K].
// mode 0: -> Q bf16 [B,H,S,D], scaled by 1/sqrt(D)
// mode 1: -> K bf16 [B,H,S,D]
// mode 2: -> V bf16 [B,H,D,S]  (transposed for the PV B-operand)
// mode 3: -> fp32 [M,N] (final output)
// Block: 256 threads (8 waves), tile 128(M) x 64(N), K-step 32.
// ---------------------------------------------------------------------------
#define LDSA 40   // 32 + 8 pad (bf16 elems) -> 80B row stride, 16B aligned
__global__ __launch_bounds__(256)
void gemm_kernel(const bf16* __restrict__ A, const bf16* __restrict__ Bt,
                 const float* __restrict__ bias, void* __restrict__ out, int mode) {
  __shared__ bf16 sA[128 * LDSA];
  __shared__ bf16 sB[64 * LDSA];
  int tid  = threadIdx.x;
  int wave = tid >> 5, lane = tid & 31;
  int m16  = lane & 15, half = lane >> 4;
  int m0 = blockIdx.x * 128;
  int n0 = blockIdx.y * 64;

  v8f acc[4] = {};

  for (int kb = 0; kb < EMB; kb += 32) {
    // Stage A tile: 128 rows x 32 cols bf16 = 512 x 16B chunks (async->LDS).
#pragma unroll
    for (int i = 0; i < 2; ++i) {
      int chunk = tid * 2 + i;
      int r = chunk >> 2, c = (chunk & 3) * 8;
      async_copy16(&A[(size_t)(m0 + r) * EMB + kb + c], &sA[r * LDSA + c]);
    }
    { // Stage B tile: 64 rows x 32 cols = 256 x 16B chunks.
      int r = tid >> 2, c = (tid & 3) * 8;
      async_copy16(&Bt[(size_t)(n0 + r) * EMB + kb + c], &sB[r * LDSA + c]);
    }
    if (kb + 32 < EMB)
      __builtin_prefetch(&A[(size_t)(m0 + (tid >> 1)) * EMB + kb + 32], 0, 1);
    async_drain();
    __syncthreads();

    v16bf a = load_a(&sA[wave * 16 * LDSA], LDSA, 0, 0);
#pragma unroll
    for (int nt = 0; nt < 4; ++nt) {
      v16bf b = load_b(&sB[nt * 16 * LDSA], LDSA, 0, 0);
      acc[nt] = wmma_bf16(a, b, acc[nt]);
    }
    __syncthreads();
  }

  // Epilogue: C layout -> VGPR j holds (row = half*8 + j, col = lane&15).
  int wm = m0 + wave * 16;
#pragma unroll
  for (int nt = 0; nt < 4; ++nt) {
#pragma unroll
    for (int j = 0; j < 8; ++j) {
      int row = wm + half * 8 + j;          // global M (= b*SEQ + s)
      int col = n0 + nt * 16 + m16;         // global N (= h*HDIM + d)
      float v = acc[nt][j] + bias[col];
      int b = row >> 11, s = row & (SEQ - 1);
      int h = col >> 6,  d = col & (HDIM - 1);
      if (mode == 0) {
        ((bf16*)out)[(((size_t)(b * HEADS + h)) * SEQ + s) * HDIM + d] = (bf16)(v * 0.125f);
      } else if (mode == 1) {
        ((bf16*)out)[(((size_t)(b * HEADS + h)) * SEQ + s) * HDIM + d] = (bf16)v;
      } else if (mode == 2) {
        ((bf16*)out)[(((size_t)(b * HEADS + h)) * HDIM + d) * SEQ + s] = (bf16)v;
      } else {
        ((float*)out)[(size_t)row * EMB + col] = v;
      }
    }
  }
}

// ---------------------------------------------------------------------------
// Flash attention: one (query-tile=64, head, batch) per block, 4 waves.
// Each wave owns 16 queries; loop over 64-key blocks with online softmax.
// Q is pre-scaled by 1/sqrt(D). O written bf16 to [B,S,E].
// ---------------------------------------------------------------------------
#define LDSK 72   // 64 + 8 pad -> 144B row stride, 16B aligned
__global__ __launch_bounds__(128)
void flash_kernel(const bf16* __restrict__ Q, const bf16* __restrict__ K,
                  const bf16* __restrict__ Vt, bf16* __restrict__ O) {
  __shared__ bf16 sK[64 * LDSK];        // [key][d]
  __shared__ bf16 sV[64 * LDSK];        // [d][key]
  __shared__ bf16 sP[4][16 * LDSK];     // per-wave P tile [query][key]

  int tid  = threadIdx.x;
  int wave = tid >> 5, lane = tid & 31;
  int m16  = lane & 15, half = lane >> 4;
  int h = blockIdx.y, b = blockIdx.z;
  int q0 = blockIdx.x * 64 + wave * 16;

  const bf16* Qh = Q  + ((size_t)(b * HEADS + h)) * SEQ * HDIM;
  const bf16* Kh = K  + ((size_t)(b * HEADS + h)) * SEQ * HDIM;
  const bf16* Vh = Vt + ((size_t)(b * HEADS + h)) * HDIM * SEQ;

  v16bf aq[2];
  aq[0] = load_a(Qh, HDIM, q0, 0);
  aq[1] = load_a(Qh, HDIM, q0, 32);

  v8f o[4] = {};
  float mrow[8], lrow[8];
#pragma unroll
  for (int j = 0; j < 8; ++j) { mrow[j] = -1e30f; lrow[j] = 0.0f; }

  for (int kb = 0; kb < SEQ; kb += 64) {
    // Stage K tile (64x64) and Vt tile (64x64): 512 x 16B chunks each,
    // via the CDNA5 async global->LDS path (ASYNCcnt).
#pragma unroll
    for (int i = 0; i < 4; ++i) {
      int chunk = tid * 4 + i;
      int r = chunk >> 3, c = (chunk & 7) * 8;
      async_copy16(&Kh[(size_t)(kb + r) * HDIM + c], &sK[r * LDSK + c]);
      async_copy16(&Vh[(size_t)r * SEQ + kb + c],    &sV[r * LDSK + c]);
    }
    async_drain();
    __syncthreads();

    // S = Q * K^T : 16(queries) x 64(keys), 2 K-steps x 4 key subtiles.
    v8f sc[4] = {};
#pragma unroll
    for (int ks = 0; ks < 2; ++ks) {
#pragma unroll
      for (int nt = 0; nt < 4; ++nt) {
        v16bf bk = load_b(&sK[nt * 16 * LDSK], LDSK, 0, ks * 32);
        sc[nt] = wmma_bf16(aq[ks], bk, sc[nt]);
      }
    }

    // Online softmax over the 16x64 score tile.
#pragma unroll
    for (int j = 0; j < 8; ++j) {
      float mx = fmaxf(fmaxf(sc[0][j], sc[1][j]), fmaxf(sc[2][j], sc[3][j]));
#pragma unroll
      for (int d = 1; d < 16; d <<= 1) mx = fmaxf(mx, __shfl_xor(mx, d, 32));
      float mn = fmaxf(mrow[j], mx);
      float alpha = __expf(mrow[j] - mn);
      mrow[j] = mn;
      float ls = 0.0f;
#pragma unroll
      for (int nt = 0; nt < 4; ++nt) {
        float p = __expf(sc[nt][j] - mn);
        sc[nt][j] = p;
        ls += p;
      }
#pragma unroll
      for (int d = 1; d < 16; d <<= 1) ls += __shfl_xor(ls, d, 32);
      lrow[j] = lrow[j] * alpha + ls;
#pragma unroll
      for (int nt = 0; nt < 4; ++nt) o[nt][j] *= alpha;
    }

    // In-wave transpose: C layout -> A layout via per-wave LDS tile (bf16).
    bf16* pl = sP[wave];
#pragma unroll
    for (int nt = 0; nt < 4; ++nt)
#pragma unroll
      for (int j = 0; j < 8; ++j)
        pl[(half * 8 + j) * LDSK + nt * 16 + m16] = (bf16)sc[nt][j];
    asm volatile("s_wait_dscnt 0" ::: "memory");   // same-wave LDS RAW fence

    // O += P * V : 16(queries) x 64(d), K dim = 64 keys.
#pragma unroll
    for (int ks = 0; ks < 2; ++ks) {
      v16bf ap = load_a(pl, LDSK, 0, ks * 32);
#pragma unroll
      for (int nt = 0; nt < 4; ++nt) {
        v16bf bv = load_b(&sV[nt * 16 * LDSK], LDSK, 0, ks * 32);
        o[nt] = wmma_bf16(ap, bv, o[nt]);
      }
    }
    __syncthreads();
  }

  // Normalize and store bf16 into [B,S,E].
#pragma unroll
  for (int nt = 0; nt < 4; ++nt) {
#pragma unroll
    for (int j = 0; j < 8; ++j) {
      int row = q0 + half * 8 + j;
      int d = nt * 16 + m16;
      float val = o[nt][j] / lrow[j];
      O[((size_t)(b * SEQ + row)) * EMB + h * HDIM + d] = (bf16)val;
    }
  }
}

// ---------------------------------------------------------------------------
// Launcher
// ---------------------------------------------------------------------------
extern "C" void kernel_launch(void* const* d_in, const int* in_sizes, int n_in,
                              void* d_out, int out_size, void* d_ws, size_t ws_size,
                              hipStream_t stream) {
  const float* X  = (const float*)d_in[0];
  const float* wq = (const float*)d_in[1];
  const float* bq = (const float*)d_in[2];
  const float* wk = (const float*)d_in[3];
  const float* bk = (const float*)d_in[4];
  const float* wv = (const float*)d_in[5];
  const float* bv = (const float*)d_in[6];
  const float* wo = (const float*)d_in[7];
  const float* bo = (const float*)d_in[8];

  // Workspace layout (bf16 elements), total ~88 MiB.
  bf16* Xb  = (bf16*)d_ws;
  bf16* wqT = Xb  + (size_t)MROWS * EMB;
  bf16* wkT = wqT + (size_t)EMB * EMB;
  bf16* wvT = wkT + (size_t)EMB * EMB;
  bf16* woT = wvT + (size_t)EMB * EMB;
  bf16* Qb  = woT + (size_t)EMB * EMB;
  bf16* Kb  = Qb  + (size_t)MROWS * EMB;
  bf16* Vtb = Kb  + (size_t)MROWS * EMB;
  bf16* Ob  = Vtb + (size_t)MROWS * EMB;

  prep_kernel<<<(MROWS * EMB + 255) / 256, 256, 0, stream>>>(
      X, wq, wk, wv, wo, Xb, wqT, wkT, wvT, woT);

  dim3 gg(MROWS / 128, EMB / 64);
  gemm_kernel<<<gg, 256, 0, stream>>>(Xb, wqT, bq, Qb, 0);
  gemm_kernel<<<gg, 256, 0, stream>>>(Xb, wkT, bk, Kb, 1);
  gemm_kernel<<<gg, 256, 0, stream>>>(Xb, wvT, bv, Vtb, 2);

  flash_kernel<<<dim3(SEQ / 64, HEADS, BATCH), 128, 0, stream>>>(Qb, Kb, Vtb, Ob);

  gemm_kernel<<<gg, 256, 0, stream>>>(Ob, woT, bo, d_out, 3);
}
